// PointNetSetAbstraction_1022202217390
// MI455X (gfx1250) — matrix operand order repaired and checked by reading, not compiled, not run on hardware
//
#include <hip/hip_runtime.h>
#include <hip/hip_bf16.h>

// Problem constants (match reference)
#define BATCH   8
#define NPTS    16384
#define NPOINT  1024
#define NSAMP   32
#define C_IN    64
#define BIGF    1e10f
#define PPT     16          // points per thread in FPS (16 * 1024 = 16384)

typedef _Float16 v16h __attribute__((ext_vector_type(16)));
typedef float    v8f  __attribute__((ext_vector_type(8)));

// ---------------------------------------------------------------------------
// Kernel 1: farthest point sampling. One workgroup per batch. Each thread owns
// 16 points: coordinates + running distance live in VGPRs, so the 1024 serial
// iterations touch no global memory; the argmax reduction carries the winning
// point's coordinates so the next centroid needs no gather either.
// Semantics faithful to reference: dist0 = mask?0:BIG, update where
// (d < dist) & mask, farthest = argmax(dist), first-index tie-break.
// ---------------------------------------------------------------------------
__global__ void __launch_bounds__(1024)
fps_kernel(const float* __restrict__ xyz, const unsigned char* __restrict__ mask,
           int* __restrict__ fps_idx)
{
    const int b   = blockIdx.x;
    const int tid = threadIdx.x;

    __shared__ float s_rv[32], s_rx[32], s_ry[32], s_rz[32];
    __shared__ int   s_ri[32];
    __shared__ float s_c[3];
    __shared__ int   s_cur;

    const float*         X = xyz  + (size_t)b * NPTS * 3;
    const unsigned char* M = mask + (size_t)b * NPTS;

    // register-resident point set: i = tid + p*1024
    float px[PPT], py[PPT], pz[PPT], pd[PPT];
    unsigned int mb = 0;
    #pragma unroll
    for (int p = 0; p < PPT; ++p) {
        const int i = tid + (p << 10);
        px[p] = X[i * 3 + 0];
        py[p] = X[i * 3 + 1];
        pz[p] = X[i * 3 + 2];
        const unsigned int m = M[i] ? 1u : 0u;
        mb |= (m << p);
        pd[p] = m ? 0.0f : BIGF;
    }

    // far0 = argmax(mask) = first valid index (block-wide min valid index)
    {
        int besti = 0x7fffffff;
        #pragma unroll
        for (int p = PPT - 1; p >= 0; --p)
            if (mb & (1u << p)) besti = tid + (p << 10);
        #pragma unroll
        for (int o = 16; o > 0; o >>= 1) {
            const int oi = __shfl_xor(besti, o, 32);
            if (oi < besti) besti = oi;
        }
        if ((tid & 31) == 0) s_ri[tid >> 5] = besti;
        __syncthreads();
        if (tid == 0) {
            int fi = s_ri[0];
            for (int w = 1; w < 32; ++w) if (s_ri[w] < fi) fi = s_ri[w];
            if (fi == 0x7fffffff) fi = 0;       // no valid point -> argmax = 0
            s_cur  = fi;
            s_c[0] = X[fi * 3 + 0];
            s_c[1] = X[fi * 3 + 1];
            s_c[2] = X[fi * 3 + 2];
        }
        __syncthreads();
    }

    for (int it = 0; it < NPOINT; ++it) {
        const int   cur = s_cur;
        const float cx = s_c[0], cy = s_c[1], cz = s_c[2];
        if (tid == 0) fps_idx[b * NPOINT + it] = cur;

        float bv = -1.0f, bx = 0.0f, by = 0.0f, bz = 0.0f;
        int   bi = 0;
        #pragma unroll
        for (int p = 0; p < PPT; ++p) {
            const float dx = px[p] - cx;
            const float dy = py[p] - cy;
            const float dz = pz[p] - cz;
            const float d  = dx * dx + dy * dy + dz * dz;
            if (d < pd[p] && (mb & (1u << p))) pd[p] = d;
            if (pd[p] > bv) {                   // ascending i => first-index ties
                bv = pd[p]; bi = tid + (p << 10);
                bx = px[p]; by = py[p]; bz = pz[p];
            }
        }
        // wave32 argmax reduce carrying coordinates (lower index wins ties)
        #pragma unroll
        for (int o = 16; o > 0; o >>= 1) {
            const float ov = __shfl_xor(bv, o, 32);
            const int   oi = __shfl_xor(bi, o, 32);
            const float ox = __shfl_xor(bx, o, 32);
            const float oy = __shfl_xor(by, o, 32);
            const float oz = __shfl_xor(bz, o, 32);
            if (ov > bv || (ov == bv && oi < bi)) {
                bv = ov; bi = oi; bx = ox; by = oy; bz = oz;
            }
        }
        const int wid = tid >> 5;
        if ((tid & 31) == 0) {
            s_rv[wid] = bv; s_ri[wid] = bi;
            s_rx[wid] = bx; s_ry[wid] = by; s_rz[wid] = bz;
        }
        __syncthreads();
        if (tid < 32) {                         // second stage: one wave, shuffles
            float fv = s_rv[tid], fx = s_rx[tid], fy = s_ry[tid], fz = s_rz[tid];
            int   fi = s_ri[tid];
            #pragma unroll
            for (int o = 16; o > 0; o >>= 1) {
                const float ov = __shfl_xor(fv, o, 32);
                const int   oi = __shfl_xor(fi, o, 32);
                const float ox = __shfl_xor(fx, o, 32);
                const float oy = __shfl_xor(fy, o, 32);
                const float oz = __shfl_xor(fz, o, 32);
                if (ov > fv || (ov == fv && oi < fi)) {
                    fv = ov; fi = oi; fx = ox; fy = oy; fz = oz;
                }
            }
            if (tid == 0) {
                s_cur = fi; s_c[0] = fx; s_c[1] = fy; s_c[2] = fz;
            }
        }
        __syncthreads();
    }
}

// ---------------------------------------------------------------------------
// Kernel 2: 32-nearest-neighbor. One WAVE per query (8192 waves). Each lane
// scans the residue class {lane, lane+32, ...} (coalesced), keeping a
// register-resident sorted-32 list; then a 32-round wave-min merge extracts
// the global nearest-first list (lower-index tie-break = top_k's ordering).
// ---------------------------------------------------------------------------
__global__ void __launch_bounds__(256)
knn_kernel(const float* __restrict__ xyz, const unsigned char* __restrict__ mask,
           const int* __restrict__ fps_idx,
           float* __restrict__ new_xyz, int* __restrict__ knn_idx)
{
    const int lane = threadIdx.x & 31;
    const int q    = (blockIdx.x * blockDim.x + threadIdx.x) >> 5;  // query id
    const int b    = q >> 10;                                       // NPOINT=1024

    const int ci = fps_idx[q];
    const float* Xb = xyz + (size_t)b * NPTS * 3;
    const unsigned char* Mb = mask + (size_t)b * NPTS;

    const float cx = Xb[ci * 3 + 0];
    const float cy = Xb[ci * 3 + 1];
    const float cz = Xb[ci * 3 + 2];
    if (lane < 3) new_xyz[q * 3 + lane] = (lane == 0) ? cx : ((lane == 1) ? cy : cz);

    float bd[NSAMP];
    int   bi[NSAMP];
    #pragma unroll
    for (int j = 0; j < NSAMP; ++j) { bd[j] = 1e30f; bi[j] = 0; }

    for (int i = lane; i < NPTS; i += 32) {     // 512 points per lane
        const float dx = Xb[i * 3 + 0] - cx;
        const float dy = Xb[i * 3 + 1] - cy;
        const float dz = Xb[i * 3 + 2] - cz;
        float d = dx * dx + dy * dy + dz * dz;
        d = Mb[i] ? d : BIGF;
        if (d < bd[NSAMP - 1]) {
            float dj = d; int ij = i;
            #pragma unroll
            for (int j = 0; j < NSAMP; ++j) {   // compare-swap chain, regs only
                if (dj < bd[j]) {
                    const float tf = bd[j]; bd[j] = dj; dj = tf;
                    const int   ti = bi[j]; bi[j] = ij; ij = ti;
                }
            }
        }
    }

    // 32-round wave merge of 32 sorted lists
    int keep = 0;
    #pragma unroll
    for (int r = 0; r < NSAMP; ++r) {
        float mv = bd[0];
        int   mi = bi[0];
        #pragma unroll
        for (int o = 16; o > 0; o >>= 1) {      // all-lanes wave-min butterfly
            const float ov = __shfl_xor(mv, o, 32);
            const int   oi = __shfl_xor(mi, o, 32);
            if (ov < mv || (ov == mv && oi < mi)) { mv = ov; mi = oi; }
        }
        if (lane == r) keep = mi;
        if ((mi & 31) == lane) {                // winner pops its head
            #pragma unroll
            for (int j = 0; j < NSAMP - 1; ++j) { bd[j] = bd[j + 1]; bi[j] = bi[j + 1]; }
            bd[NSAMP - 1] = 1e30f;
        }
    }
    knn_idx[(size_t)q * NSAMP + lane] = keep;
}

// ---------------------------------------------------------------------------
// Kernel 3: gather + 3-layer pointwise MLP (WMMA) + max over K samples.
// One wave per query; 2 waves per block. f16 inputs, f32 accumulate.
// Fragment assembly follows ISA 7.12.2 wave32 layouts.
// ---------------------------------------------------------------------------
__device__ inline v16h ldsA(const _Float16* base, int ld, int mbase, int kbase, int lane)
{
    v16h a;
    const int half = lane >> 4;
    const int m    = (lane & 15) + mbase;
    #pragma unroll
    for (int e = 0; e < 16; ++e) {
        const int v = e >> 1, w = e & 1;
        const int k = ((v & 4) ? 16 : 0) + (v & 3) * 2 + w + half * 8 + kbase;
        a[e] = base[m * ld + k];
    }
    return a;
}

__device__ inline v16h ldsB(const _Float16* wmat, int ld, int nbase, int kbase, int lane)
{
    v16h bf;
    const int half = lane >> 4;
    const int n    = (lane & 15) + nbase;
    #pragma unroll
    for (int e = 0; e < 16; ++e) {
        const int k = (e & 7) + ((e >> 3) << 4) + half * 8 + kbase;
        bf[e] = wmat[n * ld + k];
    }
    return bf;
}

template <int KT, int NT>
__device__ inline void mlp_layer(const _Float16* in, int ldin,
                                 const _Float16* w, int ldw,
                                 const float* bias,
                                 _Float16* out, int ldout, int lane)
{
    #pragma unroll
    for (int mt = 0; mt < 2; ++mt) {
        v16h af[KT];
        #pragma unroll
        for (int kt = 0; kt < KT; ++kt) af[kt] = ldsA(in, ldin, mt * 16, kt * 32, lane);
        #pragma unroll
        for (int nt = 0; nt < NT; ++nt) {
            v8f c = {};
            #pragma unroll
            for (int kt = 0; kt < KT; ++kt) {
                const v16h bf = ldsB(w, ldw, nt * 16, kt * 32, lane);
                c = __builtin_amdgcn_wmma_f32_16x16x32_f16(
                        false, af[kt], false, bf, (short)0, c, false, false);
            }
            const int half = lane >> 4;
            const int n    = (lane & 15) + nt * 16;
            const float bv = bias[n];
            #pragma unroll
            for (int v = 0; v < 8; ++v) {
                float x = c[v] + bv;
                x = x > 0.0f ? x : 0.0f;
                out[(mt * 16 + v + 8 * half) * ldout + n] = (_Float16)x;
            }
        }
    }
}

__global__ void __launch_bounds__(64)
mlp_kernel(const float* __restrict__ xyz, const float* __restrict__ points,
           const int* __restrict__ knn_idx, const float* __restrict__ new_xyz,
           const float* __restrict__ w0, const float* __restrict__ b0,
           const float* __restrict__ w1, const float* __restrict__ b1,
           const float* __restrict__ w2, const float* __restrict__ b2,
           float* __restrict__ out_pts)
{
    const int tid  = threadIdx.x;
    const int lane = tid & 31;
    const int wave = tid >> 5;
    const int q    = blockIdx.x * 2 + wave;     // global query id
    const int b    = q >> 10;                   // NPOINT = 1024

    // LDS: weights (f16, padded), biases (f32), per-wave activation buffers
    __shared__ _Float16 sW0[64 * 96];
    __shared__ _Float16 sW1[64 * 64];
    __shared__ _Float16 sW2[128 * 64];
    __shared__ float    sB0[64], sB1[64], sB2[128];
    __shared__ _Float16 sF[2][32 * 96];         // feat / layer-1 output
    __shared__ _Float16 sH[2][32 * 64];         // layer-0 output

    for (int i = tid; i < 64 * 96; i += 64) {
        const int co = i / 96, ci = i % 96;
        sW0[i] = (ci < 67) ? (_Float16)w0[co * 67 + ci] : (_Float16)0.0f;
    }
    for (int i = tid; i < 64 * 64;  i += 64) sW1[i] = (_Float16)w1[i];
    for (int i = tid; i < 128 * 64; i += 64) sW2[i] = (_Float16)w2[i];
    if (tid < 64)  { sB0[tid] = b0[tid]; sB1[tid] = b1[tid]; }
    for (int i = tid; i < 128; i += 64) sB2[i] = b2[i];

    // gather: lane j handles sample j of this query
    const int   pt = knn_idx[(size_t)q * NSAMP + lane];
    const float cx = new_xyz[q * 3 + 0];
    const float cy = new_xyz[q * 3 + 1];
    const float cz = new_xyz[q * 3 + 2];
    const float* Pp = points + ((size_t)b * NPTS + pt) * C_IN;
    const float* Xp = xyz    + ((size_t)b * NPTS + pt) * 3;
    _Float16* row = &sF[wave][lane * 96];
    row[0] = (_Float16)(Xp[0] - cx);
    row[1] = (_Float16)(Xp[1] - cy);
    row[2] = (_Float16)(Xp[2] - cz);
    #pragma unroll 8
    for (int c = 0; c < C_IN; ++c) row[3 + c] = (_Float16)Pp[c];
    #pragma unroll
    for (int c = 67; c < 96; ++c)  row[c] = (_Float16)0.0f;
    __syncthreads();

    // layer 0: (32 x 96[67]) x (96 x 64) -> sH    [24 WMMA]
    mlp_layer<3, 4>(sF[wave], 96, sW0, 96, sB0, sH[wave], 64, lane);
    __syncthreads();
    // layer 1: (32 x 64) x (64 x 64) -> sF (ld 96) [16 WMMA]
    mlp_layer<2, 4>(sH[wave], 64, sW1, 64, sB1, sF[wave], 96, lane);
    __syncthreads();

    // layer 2 + max over the 32 samples: (32 x 64) x (64 x 128) [32 WMMA]
    float cm[8];
    #pragma unroll
    for (int t = 0; t < 8; ++t) cm[t] = 0.0f;   // relu outputs are >= 0
    #pragma unroll
    for (int mt = 0; mt < 2; ++mt) {
        v16h af[2];
        af[0] = ldsA(sF[wave], 96, mt * 16, 0,  lane);
        af[1] = ldsA(sF[wave], 96, mt * 16, 32, lane);
        #pragma unroll
        for (int nt = 0; nt < 8; ++nt) {
            v8f c = {};
            #pragma unroll
            for (int kt = 0; kt < 2; ++kt) {
                const v16h bf = ldsB(sW2, 64, nt * 16, kt * 32, lane);
                c = __builtin_amdgcn_wmma_f32_16x16x32_f16(
                        false, af[kt], false, bf, (short)0, c, false, false);
            }
            const float bv = sB2[(lane & 15) + nt * 16];
            #pragma unroll
            for (int v = 0; v < 8; ++v) {
                float x = c[v] + bv;
                x = x > 0.0f ? x : 0.0f;
                cm[nt] = fmaxf(cm[nt], x);
            }
        }
    }
    #pragma unroll
    for (int nt = 0; nt < 8; ++nt) {
        const float o = fmaxf(cm[nt], __shfl_xor(cm[nt], 16, 32));
        if (lane < 16)
            out_pts[(size_t)q * 128 + nt * 16 + lane] = o;
    }
}

// ---------------------------------------------------------------------------
extern "C" void kernel_launch(void* const* d_in, const int* in_sizes, int n_in,
                              void* d_out, int out_size, void* d_ws, size_t ws_size,
                              hipStream_t stream)
{
    (void)in_sizes; (void)n_in; (void)out_size; (void)ws_size;

    const float*         xyz    = (const float*)d_in[0];
    const float*         points = (const float*)d_in[1];
    const unsigned char* mask   = (const unsigned char*)d_in[2];
    const float*         w0     = (const float*)d_in[3];
    const float*         b0     = (const float*)d_in[4];
    const float*         w1     = (const float*)d_in[5];
    const float*         b1     = (const float*)d_in[6];
    const float*         w2     = (const float*)d_in[7];
    const float*         b2     = (const float*)d_in[8];

    float* out_xyz = (float*)d_out;                                // (B,S,3)
    float* out_pts = out_xyz + (size_t)BATCH * NPOINT * 3;         // (B,S,128)

    int* fps_idx = (int*)d_ws;                                     // B*S ints
    int* knn_idx = fps_idx + BATCH * NPOINT;                       // B*S*32 ints

    fps_kernel<<<BATCH, 1024, 0, stream>>>(xyz, mask, fps_idx);

    // 8192 queries, one wave each, 8 waves per 256-thread block
    knn_kernel<<<(BATCH * NPOINT) / 8, 256, 0, stream>>>(
        xyz, mask, fps_idx, out_xyz, knn_idx);

    mlp_kernel<<<(BATCH * NPOINT) / 2, 64, 0, stream>>>(
        xyz, points, knn_idx, out_xyz, w0, b0, w1, b1, w2, b2, out_pts);
}
